// RNNSent_5334349381947
// MI455X (gfx1250) — compile-verified, matrix operand order checked by reading, not run on hardware
//
#include <hip/hip_runtime.h>
#include <hip/hip_bf16.h>

typedef __attribute__((ext_vector_type(16))) _Float16 v16h;
typedef __attribute__((ext_vector_type(8)))  float    v8f;

#define T_SEQ   300
#define NGATE   1024
#define KC0     8      // K chunks (x32) for layer-0 recurrent GEMM (K=256)
#define KC1     16     // K chunks (x32) for layer-1 fused GEMM     (K=512)

// ---------------- workspace layout (byte offsets, all 32B aligned) ------------
#define WS_XW0    0                          // float[300*4*1024]   = 4,915,200 B
#define WS_FRAG0  4915200                    // _Float16[64*8*512]  =   524,288 B
#define WS_FRAG1  (WS_FRAG0 + 524288)        // _Float16[64*16*512] = 1,048,576 B
#define WS_BIAS1  (WS_FRAG1 + 1048576)       // float[1024]
#define WS_FCOUT  (WS_BIAS1 + 4096)          // float[40]

__device__ __forceinline__ float sigf(float x) { return 1.0f / (1.0f + __expf(-x)); }

// ---------------------------------------------------------------------------
// Kernel 1: embedding gather + full input projection for layer 0 (off the
// serial chain, fully parallel).
// xw0[t][m][n] = sum_j x[t][m][j] * W_ih0[n][j] + b_ih0[n] + b_hh0[n]
// ---------------------------------------------------------------------------
__global__ void k_embed_xw0(const float* __restrict__ encoder,
                            const int*   __restrict__ inp,
                            const float* __restrict__ W_ih0,
                            const float* __restrict__ b_ih0,
                            const float* __restrict__ b_hh0,
                            float*       __restrict__ xw0)
{
    __shared__ float sx[25];
    const int tm = blockIdx.x;          // 0..1199 : (t,m) pair
    const int t  = tm >> 2;
    const int m  = tm & 3;
    const int tid = threadIdx.x;

    if (tid < 25) {
        int f = t * 100 + m * 25 + tid;         // flat idx in (25,4,300) buffer
        int s = f / 1200;
        int r = f - s * 1200;
        int b = r / 300;
        int e = r - b * 300;
        int tok = inp[b * 25 + s];              // inp is (4,25); inp.T[s][b]
        sx[tid] = encoder[(size_t)tok * 300 + e];
    }
    __syncthreads();

    for (int n = tid; n < NGATE; n += blockDim.x) {
        float acc = b_ih0[n] + b_hh0[n];
        const float* wr = W_ih0 + n * 25;
#pragma unroll
        for (int j = 0; j < 25; ++j) acc = fmaf(sx[j], wr[j], acc);
        xw0[(size_t)tm * NGATE + n] = acc;
    }
}

// ---------------------------------------------------------------------------
// Kernel 2: fp32 -> fp16 weight conversion, pre-swizzled into the CDNA5
// 16-bit B-matrix (32x16) WMMA fragment layout, fragment-major so each lane
// loads its 16 halves as one contiguous 32B v16h.
//   frag0 : W_hh0            (K=256, 8 chunks / 64 n-tiles)
//   frag1 : [W_ih1 ; W_hh1]  (K=512, 16 chunks / 64 n-tiles)  (B[k][n]=W[n][k])
// Also builds bias1 = b_ih1 + b_hh1.
// ---------------------------------------------------------------------------
__global__ void k_swizzle(const float* __restrict__ W_hh0,
                          const float* __restrict__ W_ih1,
                          const float* __restrict__ W_hh1,
                          const float* __restrict__ b_ih1,
                          const float* __restrict__ b_hh1,
                          _Float16*    __restrict__ frag0,
                          _Float16*    __restrict__ frag1,
                          float*       __restrict__ bias1)
{
    const int g = blockIdx.x * blockDim.x + threadIdx.x;
    const int N0 = 64 * KC0 * 512;   // 262144
    const int N1 = 64 * KC1 * 512;   // 524288

    if (g < N0) {
        int fragIdx = g >> 9, e = g & 511;
        int ntile = fragIdx >> 3, kchunk = fragIdx & 7;
        int lane = e >> 4, i = e & 15;
        int v = i >> 1, jj = i & 1;
        int nn = lane & 15, hi = lane >> 4;
        int k = kchunk * 32 + ((v & 4) << 2) + (hi << 3) + ((v & 3) << 1) + jj;
        int n = ntile * 16 + nn;
        frag0[g] = (_Float16)W_hh0[n * 256 + k];
    } else if (g < N0 + N1) {
        int g2 = g - N0;
        int fragIdx = g2 >> 9, e = g2 & 511;
        int ntile = fragIdx >> 4, kchunk = fragIdx & 15;
        int lane = e >> 4, i = e & 15;
        int v = i >> 1, jj = i & 1;
        int nn = lane & 15, hi = lane >> 4;
        int k = kchunk * 32 + ((v & 4) << 2) + (hi << 3) + ((v & 3) << 1) + jj;
        int n = ntile * 16 + nn;
        float val = (k < 256) ? W_ih1[n * 256 + k] : W_hh1[n * 256 + (k - 256)];
        frag1[g2] = (_Float16)val;
    } else if (g < N0 + N1 + NGATE) {
        int n = g - N0 - N1;
        bias1[n] = b_ih1[n] + b_hh1[n];
    }
}

// ---------------------------------------------------------------------------
// Dual-tile WMMA GEMM: one shared A fragment per K-chunk feeds two
// independent accumulator chains (tiles nt0, nt0+1).
// ---------------------------------------------------------------------------
__device__ __forceinline__ void gemm_pair(const _Float16* A_lds, int kchunks,
                                          const _Float16* frag, int nt0, int lane,
                                          v8f& acc0, v8f& acc1)
{
    const int mm = lane & 15;
    const int hi = lane >> 4;
    const _Float16* Arow = A_lds + mm * 512;
    const v16h* bp = (const v16h*)frag;
#pragma unroll
    for (int kc = 0; kc < kchunks; ++kc) {
        const _Float16* ar = Arow + kc * 32;
        v16h a;
#pragma unroll
        for (int v = 0; v < 8; ++v) {
            const int kb = ((v & 4) << 2) + (hi << 3) + ((v & 3) << 1);
            a[2 * v]     = ar[kb];
            a[2 * v + 1] = ar[kb + 1];
        }
        v16h b0 = bp[(size_t)(nt0 * kchunks + kc) * 32 + lane];
        v16h b1 = bp[(size_t)((nt0 + 1) * kchunks + kc) * 32 + lane];
        acc0 = __builtin_amdgcn_wmma_f32_16x16x32_f16(false, a, false, b0,
                                                      (short)0, acc0, false, false);
        acc1 = __builtin_amdgcn_wmma_f32_16x16x32_f16(false, a, false, b1,
                                                      (short)0, acc1, false, false);
    }
}

// ---------------------------------------------------------------------------
// Kernel 3: persistent pipelined 2-layer LSTM. One workgroup = 32 wave32.
// Wavefront pipelining: at iteration t one fused GEMM phase computes BOTH
//   gates0(t)   = xw0[t]  + h0[t-1] @ W_hh0^T                 (K=256)
//   gates1(t-1) = bias1   + [y0[t-1]; h1[t-2]] @ [Wih1;Whh1]^T (K=512)
// (y0[t-1] == h0[t-1]: same 256 LDS columns feed both GEMMs), then one merged
// pointwise phase updates both layers. 2 barriers/step instead of 4, and each
// wave issues its 48 WMMAs back-to-back. Uniform t==0 / t==300 fill/drain
// branches keep EXEC all-ones around WMMA.
// ---------------------------------------------------------------------------
__global__ void __launch_bounds__(1024, 1)
k_recurrent(const float*    __restrict__ xw0,
            const _Float16* __restrict__ frag0,
            const _Float16* __restrict__ frag1,
            const float*    __restrict__ bias1,
            const float*    __restrict__ h_init,
            const float*    __restrict__ c_init,
            const float*    __restrict__ fc_W,
            const float*    __restrict__ fc_b,
            float*          __restrict__ fc_out,
            float*          __restrict__ out_hc)
{
    __shared__ _Float16 A_lds[16 * 512];     // cols 0..255: h0/y0, 256..511: h1
    __shared__ float    gates0[4 * NGATE];
    __shared__ float    gates1[4 * NGATE];
    __shared__ float    h_state[2][4 * 256];
    __shared__ float    c_state[2][4 * 256];

    const int tid  = threadIdx.x;
    const int lane = tid & 31;
    const int wave = tid >> 5;
    const int nt0  = wave * 2;
    const int m    = tid >> 8;     // batch row 0..3
    const int j    = tid & 255;    // hidden unit

    for (int idx = tid; idx < 16 * 512; idx += 1024) A_lds[idx] = (_Float16)0.0f;
    __syncthreads();
    {
        float h0v = h_init[0 * 1024 + m * 256 + j];
        float h1v = h_init[1 * 1024 + m * 256 + j];
        h_state[0][m * 256 + j] = h0v;
        h_state[1][m * 256 + j] = h1v;
        c_state[0][m * 256 + j] = c_init[0 * 1024 + m * 256 + j];
        c_state[1][m * 256 + j] = c_init[1 * 1024 + m * 256 + j];
        A_lds[m * 512 + j]       = (_Float16)h0v;
        A_lds[m * 512 + 256 + j] = (_Float16)h1v;
    }
    __syncthreads();

    // loop-invariant C-init values
    const int   col  = lane & 15;
    const float bv0  = bias1[nt0 * 16 + col];
    const float bv1  = bias1[(nt0 + 1) * 16 + col];

    for (int t = 0; t <= T_SEQ; ++t) {
        // Opaque offset: stops LICM from hoisting (and spilling) the 48
        // loop-invariant B fragments; they stream from L2 every step instead.
        int bofs = 0;
        asm volatile("" : "+s"(bofs));
        const _Float16* f0 = frag0 + bofs;
        const _Float16* f1 = frag1 + bofs;

        // ---- fused GEMM phase ----
        if (t < T_SEQ) {                             // layer 0, timestep t
            const float* xwt = xw0 + (size_t)t * 4096;
            v8f a0, a1;
#pragma unroll
            for (int r = 0; r < 8; ++r) { a0[r] = 0.0f; a1[r] = 0.0f; }
#pragma unroll
            for (int r = 0; r < 4; ++r) {            // rows 8..15 get dup junk;
                a0[r] = xwt[r * NGATE + nt0 * 16 + col];        // ignored
                a1[r] = xwt[r * NGATE + (nt0 + 1) * 16 + col];
            }
            gemm_pair(A_lds, KC0, f0, nt0, lane, a0, a1);
            if (lane < 16) {
#pragma unroll
                for (int r = 0; r < 4; ++r) {
                    gates0[r * NGATE + nt0 * 16 + lane]       = a0[r];
                    gates0[r * NGATE + (nt0 + 1) * 16 + lane] = a1[r];
                }
            }
            if (t + 1 < T_SEQ && tid < 128)          // pull next C-init to L2/L0
                __builtin_prefetch(xw0 + (size_t)(t + 1) * 4096 + tid * 32, 0, 3);
        }
        if (t > 0) {                                 // layer 1, timestep t-1
            v8f a0, a1;
#pragma unroll
            for (int r = 0; r < 8; ++r) {
                a0[r] = (r < 4) ? bv0 : 0.0f;
                a1[r] = (r < 4) ? bv1 : 0.0f;
            }
            gemm_pair(A_lds, KC1, f1, nt0, lane, a0, a1);
            if (lane < 16) {
#pragma unroll
                for (int r = 0; r < 4; ++r) {
                    gates1[r * NGATE + nt0 * 16 + lane]       = a0[r];
                    gates1[r * NGATE + (nt0 + 1) * 16 + lane] = a1[r];
                }
            }
        }
        __syncthreads();

        // ---- merged pointwise phase ----
        if (t < T_SEQ) {
            float gi = gates0[m * NGATE + j];
            float gf = gates0[m * NGATE + 256 + j];
            float gg = gates0[m * NGATE + 512 + j];
            float go = gates0[m * NGATE + 768 + j];
            float c  = sigf(gf) * c_state[0][m * 256 + j] + sigf(gi) * tanhf(gg);
            float h  = sigf(go) * tanhf(c);
            c_state[0][m * 256 + j] = c;
            h_state[0][m * 256 + j] = h;
            A_lds[m * 512 + j] = (_Float16)h;        // h0[t] == y0[t]
        }
        if (t > 0) {
            float gi = gates1[m * NGATE + j];
            float gf = gates1[m * NGATE + 256 + j];
            float gg = gates1[m * NGATE + 512 + j];
            float go = gates1[m * NGATE + 768 + j];
            float c  = sigf(gf) * c_state[1][m * 256 + j] + sigf(gi) * tanhf(gg);
            float h  = sigf(go) * tanhf(c);
            c_state[1][m * 256 + j] = c;
            h_state[1][m * 256 + j] = h;
            A_lds[m * 512 + 256 + j] = (_Float16)h;  // h1[t-1]
        }
        __syncthreads();
    }

    // ---- hidden/cell outputs: (2,4,256) h then (2,4,256) c ----
    out_hc[0 * 1024 + m * 256 + j]        = h_state[0][m * 256 + j];
    out_hc[1 * 1024 + m * 256 + j]        = h_state[1][m * 256 + j];
    out_hc[2048 + 0 * 1024 + m * 256 + j] = c_state[0][m * 256 + j];
    out_hc[2048 + 1 * 1024 + m * 256 + j] = c_state[1][m * 256 + j];

    // ---- FC on last-step y1: out[m][k] = relu(y1 . fc_W[k] + fc_b[k]) ----
    if (tid < 40) {
        const int mm = tid / 10, k = tid % 10;
        float s = fc_b[k];
        for (int jj = 0; jj < 256; ++jj)
            s = fmaf(h_state[1][mm * 256 + jj], fc_W[k * 256 + jj], s);
        fc_out[tid] = fmaxf(s, 0.0f);    // fc_out[mm*10+k]
    }
}

// ---------------------------------------------------------------------------
// Kernel 4: decoder GEMV (the only HBM-streaming stage: ~5.6MB @ 23.3 TB/s).
// decoded.T layout: d_out[v*4 + m].
// ---------------------------------------------------------------------------
__global__ void k_decode(const float* __restrict__ fc_out,
                         const float* __restrict__ dec_W,
                         const float* __restrict__ dec_b,
                         float*       __restrict__ out)
{
    __shared__ float sfc[40];
    const int tid = threadIdx.x;
    if (tid < 40) sfc[tid] = fc_out[tid];
    __syncthreads();

    const int v = blockIdx.x * blockDim.x + tid;
    if (v >= 100000) return;

    float w[10];
#pragma unroll
    for (int k = 0; k < 10; ++k) w[k] = dec_W[(size_t)v * 10 + k];
    const float bb = dec_b[v];

    float4 r;
    float* rp = &r.x;
#pragma unroll
    for (int mm = 0; mm < 4; ++mm) {
        float s = bb;
#pragma unroll
        for (int k = 0; k < 10; ++k) s = fmaf(sfc[mm * 10 + k], w[k], s);
        rp[mm] = s;
    }
    *(float4*)(out + (size_t)v * 4) = r;
}

// ---------------------------------------------------------------------------
extern "C" void kernel_launch(void* const* d_in, const int* in_sizes, int n_in,
                              void* d_out, int out_size, void* d_ws, size_t ws_size,
                              hipStream_t stream)
{
    const float* encoder = (const float*)d_in[0];
    const float* h0      = (const float*)d_in[1];
    const float* c0      = (const float*)d_in[2];
    const float* W_ih0   = (const float*)d_in[3];
    const float* W_hh0   = (const float*)d_in[4];
    const float* b_ih0   = (const float*)d_in[5];
    const float* b_hh0   = (const float*)d_in[6];
    const float* W_ih1   = (const float*)d_in[7];
    const float* W_hh1   = (const float*)d_in[8];
    const float* b_ih1   = (const float*)d_in[9];
    const float* b_hh1   = (const float*)d_in[10];
    const float* fc_W    = (const float*)d_in[11];
    const float* fc_b    = (const float*)d_in[12];
    const float* dec_W   = (const float*)d_in[13];
    const float* dec_b   = (const float*)d_in[14];
    const int*   inp     = (const int*)d_in[15];
    float* out = (float*)d_out;

    char* ws = (char*)d_ws;
    float*    xw0    = (float*)(ws + WS_XW0);
    _Float16* frag0  = (_Float16*)(ws + WS_FRAG0);
    _Float16* frag1  = (_Float16*)(ws + WS_FRAG1);
    float*    bias1  = (float*)(ws + WS_BIAS1);
    float*    fc_out = (float*)(ws + WS_FCOUT);

    // Phase A: fully parallel prep (off the serial chain).
    k_embed_xw0<<<1200, 256, 0, stream>>>(encoder, inp, W_ih0, b_ih0, b_hh0, xw0);
    k_swizzle<<<3076, 256, 0, stream>>>(W_hh0, W_ih1, W_hh1, b_ih1, b_hh1,
                                        frag0, frag1, bias1);

    // Phase B: latency-bound recurrence, one persistent WGP, WMMA per step.
    k_recurrent<<<1, 1024, 0, stream>>>(xw0, frag0, frag1, bias1, h0, c0,
                                        fc_W, fc_b, fc_out, out + 400000);

    // Phase C: bandwidth-bound decoder GEMV.
    k_decode<<<391, 256, 0, stream>>>(fc_out, dec_W, dec_b, out);
}
